// MultiHeadAttention_13761075216785
// MI455X (gfx1250) — matrix operand order repaired and checked by reading, not compiled, not run on hardware
//
#include <hip/hip_runtime.h>
#include <stdint.h>

typedef unsigned short u16t;
typedef __attribute__((ext_vector_type(8)))  unsigned short us8;   // 16 bytes
typedef __attribute__((ext_vector_type(16))) __bf16        bf16x16;
typedef __attribute__((ext_vector_type(8)))  float         v8f;
typedef __attribute__((ext_vector_type(4)))  uint32_t      u32x4;
typedef __attribute__((ext_vector_type(8)))  uint32_t      u32x8;

#define DEV __device__ __forceinline__

DEV u16t f2bf(float f) {
    union { float f; unsigned u; } x; x.f = f;
    unsigned r = x.u + 0x7FFFu + ((x.u >> 16) & 1u);   // round-to-nearest-even
    return (u16t)(r >> 16);
}

// ---- WMMA 16x16x32 bf16 fragment loaders (ISA 7.12.2 layouts) ----
// A (16x32, MxK): lane m=lane&15, half=lane>>4.
// elements 0..7  = A[m, k0+8*half   .. +8)
// elements 8..15 = A[m, k0+16+8*half.. +8)
DEV bf16x16 load_a_bf16(const u16t* A, int lda, int row, int k0, int lane) {
    const int half = lane >> 4;
    const u16t* p = A + (size_t)row * lda + k0 + 8 * half;
    union { bf16x16 v; us8 c[2]; } u;
    u.c[0] = *(const us8*)(p);
    u.c[1] = *(const us8*)(p + 16);
    return u.v;
}

// A fragment sourced from an f32 tile staged in LDS (32 floats per row)
DEV bf16x16 load_a_lds_f32(const float* tile, int row_local, int lane) {
    const int half = lane >> 4;
    const float* p = tile + row_local * 32 + 8 * half;
    union { bf16x16 v; u16t h[16]; } u;
#pragma unroll
    for (int i = 0; i < 8; ++i) u.h[i] = f2bf(p[i]);
#pragma unroll
    for (int i = 0; i < 8; ++i) u.h[8 + i] = f2bf(p[16 + i]);
    return u.v;
}

// B (32x16, KxN) from Bm stored [N, K] row-major: lane n=lane&15, half=lane>>4,
// elements 0..7 = Bm[n, k0+8*half .. +8), elements 8..15 = Bm[n, k0+16+8*half .. +8)
DEV bf16x16 load_b_bf16(const u16t* B, int ldb, int col, int k0, int lane) {
    const int half = lane >> 4;
    const u16t* p = B + (size_t)col * ldb + k0 + 8 * half;
    union { bf16x16 v; us8 c[2]; } u;
    u.c[0] = *(const us8*)(p);
    u.c[1] = *(const us8*)(p + 16);
    return u.v;
}

template<int MODE> struct GemmDims;   // forward doc only

// MODE: 0=Q proj, 1=K proj, 2=V proj (transposed store), 3=QK^T scores,
//       4=W*V (A = f32 softmax weights, staged via Tensor Data Mover),
//       5=output proj (+bo)
template<int MODE>
__global__ __launch_bounds__(256)
void gemm_wmma(const void* A_, const u16t* Bm, const float* bias,
               void* Out_, float bscale)
{
    constexpr int S = 2048, DH = 64, H = 16, D = 1024;
    constexpr int  KDIM   = (MODE == 3) ? 64 : (MODE == 4 ? 2048 : 1024);
    constexpr int  LDA    = (MODE == 3) ? 64 : 1024;
    constexpr int  LDB    = (MODE == 3) ? 64 : (MODE == 4 ? 2048 : 1024);
    constexpr int  WAVES_M = (MODE == 4) ? 8 : 4;       // 8 waves per WG
    constexpr int  WAVES_N = 8 / WAVES_M;

    const int lane = threadIdx.x & 31;
    const int wave = threadIdx.x >> 5;
    const int wm = wave % WAVES_M;
    const int wn = wave / WAVES_M;
    const int m_wave = blockIdx.x * (WAVES_M * 32) + wm * 32;   // wave tile: 32 x 64
    const int n_wave = blockIdx.y * (WAVES_N * 64) + wn * 64;
    const int bh = blockIdx.z;

    const u16t*  Ab = (const u16t*)A_;
    const float* Af = (const float*)A_;
    if constexpr (MODE == 3) { Ab += (size_t)bh * S * DH; Bm += (size_t)bh * S * DH; }
    if constexpr (MODE == 4) { Af += (size_t)bh * S * S;  Bm += (size_t)bh * DH * S; }

    v8f acc[2][4] = {};
    const int rowA0 = m_wave + (lane & 15);
    const int colB0 = n_wave + (lane & 15);

    if constexpr (MODE == 4) {
        // ---- Tensor Data Mover staging of the f32 softmax-weight slab ----
        // WG consumes rows [blockIdx.x*256, +256), streamed 32 K-columns at a
        // time: a 256x32 f32 tile (32KB) per step, fetched by one TDM
        // descriptor issued from wave 0, tracked on TENSORcnt.
        __shared__ float tile[256 * 32];
        const uint64_t gbase =
            (uint64_t)(uintptr_t)(Af + (size_t)blockIdx.x * 256 * (size_t)S);
        const uint32_t lds_base = (uint32_t)(uintptr_t)(&tile[0]);

        // D# group 1 (invariant): data_size=4B, tensor_dim0=2048,
        // tensor_dim1=2048, tile_dim0=32, tile_dim1=256, stride0=2048.
        const u32x8 g1 = { 0x20000u,
                           (uint32_t)S << 16,
                           (uint32_t)S << 16,
                           32u << 16,
                           256u,
                           (uint32_t)S,
                           0u, 0u };

        const int row_local0 = wm * 32 + (lane & 15);
        for (int k0 = 0; k0 < KDIM; k0 += 32) {
            __syncthreads();                       // WAR on LDS tile
            if (threadIdx.x < 32) {
                const uint64_t ga = gbase + (uint64_t)k0 * 4u;
                const u32x4 g0 = { 1u,             // count=1, user descriptor
                                   lds_base,
                                   (uint32_t)ga,
                                   (uint32_t)(ga >> 32) | 0x80000000u }; // type=2
                asm volatile("tensor_load_to_lds %0, %1"
                             :: "s"(g0), "s"(g1) : "memory");
                __builtin_amdgcn_s_wait_tensorcnt(0);
            }
            __syncthreads();                       // tile visible to all waves

            bf16x16 a[2], b[4];
#pragma unroll
            for (int i = 0; i < 2; ++i)
                a[i] = load_a_lds_f32(tile, row_local0 + 16 * i, lane);
#pragma unroll
            for (int j = 0; j < 4; ++j)
                b[j] = load_b_bf16(Bm, LDB, colB0 + 16 * j, k0, lane);
            __builtin_prefetch(Bm + (size_t)colB0 * LDB + k0 + 32, 0, 1);
#pragma unroll
            for (int i = 0; i < 2; ++i)
#pragma unroll
                for (int j = 0; j < 4; ++j)
                    acc[i][j] = __builtin_amdgcn_wmma_f32_16x16x32_bf16(
                        false, a[i], false, b[j], (short)0, acc[i][j], false, false);
        }
    } else {
        for (int k0 = 0; k0 < KDIM; k0 += 32) {
            bf16x16 a[2], b[4];
#pragma unroll
            for (int i = 0; i < 2; ++i)
                a[i] = load_a_bf16(Ab, LDA, rowA0 + 16 * i, k0, lane);
#pragma unroll
            for (int j = 0; j < 4; ++j)
                b[j] = load_b_bf16(Bm, LDB, colB0 + 16 * j, k0, lane);
            __builtin_prefetch(Ab + (size_t)rowA0 * LDA + k0 + 32, 0, 1);
#pragma unroll
            for (int i = 0; i < 2; ++i)
#pragma unroll
                for (int j = 0; j < 4; ++j)
                    acc[i][j] = __builtin_amdgcn_wmma_f32_16x16x32_bf16(
                        false, a[i], false, b[j], (short)0, acc[i][j], false, false);
        }
    }

    // Epilogue. C/D layout: lane n=lane&15, VGPR r -> M = r + 8*(lane>>4).
#pragma unroll
    for (int i = 0; i < 2; ++i) {
#pragma unroll
        for (int j = 0; j < 4; ++j) {
#pragma unroll
            for (int r = 0; r < 8; ++r) {
                const int mm = m_wave + 16 * i + r + 8 * (lane >> 4);
                const int nn = n_wave + 16 * j + (lane & 15);
                float v = acc[i][j][r];
                if constexpr (MODE == 0 || MODE == 2 || MODE == 5) v += bias[nn] * bscale;
                if constexpr (MODE == 0 || MODE == 1) {
                    // store q/k as bf16 [B,H,S,dh]
                    const int b_ = mm >> 11, s_ = mm & 2047, h_ = nn >> 6, d_ = nn & 63;
                    ((u16t*)Out_)[((size_t)((b_ * H + h_) * S + s_)) * DH + d_] = f2bf(v);
                } else if constexpr (MODE == 2) {
                    // store v transposed bf16 [B,H,dh,S]
                    const int b_ = mm >> 11, s_ = mm & 2047, h_ = nn >> 6, d_ = nn & 63;
                    ((u16t*)Out_)[((size_t)((b_ * H + h_) * DH + d_)) * S + s_] = f2bf(v);
                } else if constexpr (MODE == 3) {
                    // raw attention scores f32 [B,H,S,S] (softmaxed in place later)
                    ((float*)Out_)[(size_t)bh * S * S + (size_t)mm * S + nn] = v;
                } else if constexpr (MODE == 4) {
                    // wv bf16 merged back to [B,S,D]
                    const int b_ = bh >> 4, h_ = bh & 15;
                    ((u16t*)Out_)[((size_t)(b_ * S + mm)) * D + h_ * DH + nn] = f2bf(v);
                } else {
                    ((float*)Out_)[(size_t)mm * D + nn] = v;   // final out f32 [B,S,D]
                }
            }
        }
    }
}

__global__ void cvt_kernel(const float* in, u16t* out, int n, float scale) {
    const int i = blockIdx.x * blockDim.x + threadIdx.x;
    if (i < n) out[i] = f2bf(in[i] * scale);
}

// one 2048-wide row per 256-thread block; row staged to LDS with
// GLOBAL_LOAD_ASYNC_TO_LDS_B128 (ASYNCcnt), then normalized in place (f32).
__global__ __launch_bounds__(256)
void softmax_kernel(float* w) {
    __shared__ float tile[2048];
    __shared__ float red[256];
    const int t = threadIdx.x;
    float* p = w + (size_t)blockIdx.x * 2048;

    {
        const uint32_t lds0 = (uint32_t)(uintptr_t)(&tile[t * 8]);
        const uint64_t g0   = (uint64_t)(uintptr_t)(p + t * 8);
        asm volatile(
            "global_load_async_to_lds_b128 %0, %1, off\n\t"
            "global_load_async_to_lds_b128 %0, %1, off offset:16\n\t"
            "s_wait_asynccnt 0x0"
            :: "v"(lds0), "v"(g0) : "memory");
    }
    __syncthreads();

    float x[8];
#pragma unroll
    for (int i = 0; i < 8; ++i) x[i] = tile[t + 256 * i];
    float m = x[0];
#pragma unroll
    for (int i = 1; i < 8; ++i) m = fmaxf(m, x[i]);
    red[t] = m; __syncthreads();
    for (int s = 128; s > 0; s >>= 1) { if (t < s) red[t] = fmaxf(red[t], red[t + s]); __syncthreads(); }
    m = red[0]; __syncthreads();
    float sum = 0.f;
#pragma unroll
    for (int i = 0; i < 8; ++i) { x[i] = __expf(x[i] - m); sum += x[i]; }
    red[t] = sum; __syncthreads();
    for (int s = 128; s > 0; s >>= 1) { if (t < s) red[t] += red[t + s]; __syncthreads(); }
    const float inv = 1.0f / red[0];
#pragma unroll
    for (int i = 0; i < 8; ++i) p[t + 256 * i] = x[i] * inv;
}

extern "C" void kernel_launch(void* const* d_in, const int* in_sizes, int n_in,
                              void* d_out, int out_size, void* d_ws, size_t ws_size,
                              hipStream_t stream) {
    const float* x  = (const float*)d_in[0];
    const float* Wq = (const float*)d_in[1];
    const float* bq = (const float*)d_in[2];
    const float* Wk = (const float*)d_in[3];
    const float* Wv = (const float*)d_in[4];
    const float* bv = (const float*)d_in[5];
    const float* Wo = (const float*)d_in[6];
    const float* bo = (const float*)d_in[7];

    float* out = (float*)d_out;                       // [2,2048,1024]
    float* w   = out + (size_t)2 * 2048 * 1024;       // [2,16,2048,2048]

    u16t* xb  = (u16t*)d_ws;                          // x   bf16 [4096,1024]
    u16t* wqb = xb  + (size_t)4194304;                // Wq*scale bf16 [1024,1024]
    u16t* wkb = wqb + (size_t)1048576;
    u16t* wvb = wkb + (size_t)1048576;
    u16t* wob = wvb + (size_t)1048576;
    u16t* qb  = wob + (size_t)1048576;                // q bf16 [B,H,S,dh] (pre-scaled)
    u16t* kb  = qb  + (size_t)4194304;                // k bf16 [B,H,S,dh] (pre-scaled)
    u16t* vT  = kb  + (size_t)4194304;                // v bf16 [B,H,dh,S]
    u16t* wvb_out = vT + (size_t)4194304;             // wv bf16 [B,S,D]

    const float scale = 0.35355339059327373f;         // 64^-0.25

    cvt_kernel<<<16384, 256, 0, stream>>>(x,  xb,  4194304, 1.0f);
    cvt_kernel<<<4096,  256, 0, stream>>>(Wq, wqb, 1048576, scale);
    cvt_kernel<<<4096,  256, 0, stream>>>(Wk, wkb, 1048576, scale);
    cvt_kernel<<<4096,  256, 0, stream>>>(Wv, wvb, 1048576, 1.0f);
    cvt_kernel<<<4096,  256, 0, stream>>>(Wo, wob, 1048576, 1.0f);

    dim3 gProj(32, 8, 1);   // 4096/128, 1024/128
    gemm_wmma<0><<<gProj, 256, 0, stream>>>(xb, wqb, bq, qb, scale);
    gemm_wmma<1><<<gProj, 256, 0, stream>>>(xb, wkb, nullptr, kb, 0.f);
    gemm_wmma<2><<<gProj, 256, 0, stream>>>(xb, wvb, bv, vT, 1.0f);

    dim3 gQK(16, 16, 32);   // 2048/128, 2048/128, B*H
    gemm_wmma<3><<<gQK, 256, 0, stream>>>(qb, kb, nullptr, w, 0.f);

    softmax_kernel<<<65536, 256, 0, stream>>>(w);

    dim3 gWV(8, 1, 32);     // 2048/256, 64/64, B*H
    gemm_wmma<4><<<gWV, 256, 0, stream>>>(w, vT, nullptr, wvb_out, 0.f);

    dim3 gOut(32, 8, 1);
    gemm_wmma<5><<<gOut, 256, 0, stream>>>(wvb_out, wob, bo, out, 1.0f);
}
